// _GECTLayer_48455821033934
// MI455X (gfx1250) — compile-verified, matrix operand order checked by reading, not run on hardware
//
#include <hip/hip_runtime.h>
#include <hip/hip_bf16.h>
#include <math.h>

typedef __attribute__((ext_vector_type(16))) _Float16 v16h;
typedef __attribute__((ext_vector_type(4)))  _Float16 v4h;
typedef __attribute__((ext_vector_type(8)))  float    v8f;

#define NN 50000
#define EE 500000
#define HID 128
#define NROWT (NN/16)      // 3125
#define NEDGET (EE/16)     // 31250

__device__ __forceinline__ float gelu_erf(float x) {
    return 0.5f * x * (1.0f + erff(x * 0.70710678118654752f));
}

// A fragment: 16x32 f16, row-major source with leading dim ld (contiguous 8-half chunks -> b128)
__device__ __forceinline__ v16h load_a_frag_f16(const _Float16* base, int ld, int row0, int k0) {
    int l = threadIdx.x & 31;
    int hsel = l >> 4;
    const _Float16* p = base + (size_t)(row0 + (l & 15)) * ld + k0 + hsel * 8;
    v16h a;
#pragma unroll
    for (int i = 0; i < 8; i++) a[i] = p[i];
#pragma unroll
    for (int i = 0; i < 8; i++) a[8 + i] = p[16 + i];
    return a;
}

// B fragment from pre-packed f16 buffer: 16 contiguous halfs per lane (2x b128)
__device__ __forceinline__ v16h load_b_packed(const _Float16* PB, int ct, int kt, int ktiles) {
    const _Float16* p = PB + ((size_t)(ct * ktiles + kt) * 32 + (threadIdx.x & 31)) * 16;
    v16h b;
#pragma unroll
    for (int i = 0; i < 16; i++) b[i] = p[i];
    return b;
}

__device__ __forceinline__ v8f wmma16(v16h a, v16h b, v8f c) {
    return __builtin_amdgcn_wmma_f32_16x16x32_f16(false, a, false, b, (short)0, c, false, false);
}

// -------------------- weight pre-pack: f32 W[K][Ncols] -> f16 fragment order --------------------
// grid.x = colTilesPad * (Kdim/32), blockDim = 32
__global__ void k_pack_w(const float* __restrict__ W, _Float16* __restrict__ out,
                         int Kdim, int Ncols) {
    int ktiles = Kdim >> 5;
    int ct = blockIdx.x / ktiles;
    int kt = blockIdx.x % ktiles;
    int l = threadIdx.x & 31;
    int n  = ct * 16 + (l & 15);
    int kb = kt * 32 + (l >> 4) * 8;
    _Float16* o = out + ((size_t)blockIdx.x * 32 + l) * 16;
#pragma unroll
    for (int i = 0; i < 8; i++) {
        o[i]     = (n < Ncols) ? (_Float16)W[(size_t)(kb + i) * Ncols + n]      : (_Float16)0.0f;
        o[8 + i] = (n < Ncols) ? (_Float16)W[(size_t)(kb + 16 + i) * Ncols + n] : (_Float16)0.0f;
    }
}

// -------------------- bulk f32 -> f16 convert (n divisible by 4) --------------------
__global__ void k_cvt_f16(const float* __restrict__ in, _Float16* __restrict__ out, int n4) {
    int t = blockIdx.x * blockDim.x + threadIdx.x;
    if (t >= n4) return;
    float4 v = ((const float4*)in)[t];
    v4h o;
    o[0] = (_Float16)v.x; o[1] = (_Float16)v.y; o[2] = (_Float16)v.z; o[3] = (_Float16)v.w;
    ((v4h*)out)[t] = o;
}

// -------------------- LayerNorm -> f16 --------------------
__global__ void k_ln_f16(const float* __restrict__ x, const float* __restrict__ g,
                         const float* __restrict__ b, _Float16* __restrict__ out, int n) {
    int row = blockIdx.x * (blockDim.x >> 5) + (threadIdx.x >> 5);
    if (row >= n) return;
    int l = threadIdx.x & 31;
    const float4 v = ((const float4*)(x + (size_t)row * HID))[l];
    float s = v.x + v.y + v.z + v.w;
#pragma unroll
    for (int off = 16; off; off >>= 1) s += __shfl_xor(s, off, 32);
    float mean = s * (1.0f / 128.0f);
    float d0 = v.x - mean, d1 = v.y - mean, d2 = v.z - mean, d3 = v.w - mean;
    float vs = d0 * d0 + d1 * d1 + d2 * d2 + d3 * d3;
#pragma unroll
    for (int off = 16; off; off >>= 1) vs += __shfl_xor(vs, off, 32);
    float rstd = rsqrtf(vs * (1.0f / 128.0f) + 1e-5f);
    int c = l * 4;
    v4h o;
    o[0] = (_Float16)(d0 * rstd * g[c + 0] + b[c + 0]);
    o[1] = (_Float16)(d1 * rstd * g[c + 1] + b[c + 1]);
    o[2] = (_Float16)(d2 * rstd * g[c + 2] + b[c + 2]);
    o[3] = (_Float16)(d3 * rstd * g[c + 3] + b[c + 3]);
    *(v4h*)(out + (size_t)row * HID + c) = o;
}

// -------------------- QKV projection: wave = 1 rowTile x 4 colTiles --------------------
__global__ void k_qkv(const _Float16* __restrict__ xn,
                      const _Float16* __restrict__ Wqp, const _Float16* __restrict__ Wkp,
                      const _Float16* __restrict__ Wvp,
                      float* __restrict__ Q, float* __restrict__ K, float* __restrict__ V) {
    int rt = blockIdx.x * (blockDim.x >> 5) + (threadIdx.x >> 5);
    if (rt >= NROWT) return;
    int g = blockIdx.y;                 // 0..5
    int which = g >> 1;
    int ctbase = (g & 1) * 4;
    const _Float16* Wp = (which == 0) ? Wqp : (which == 1) ? Wkp : Wvp;
    float* Out = (which == 0) ? Q : (which == 1) ? K : V;
    __builtin_prefetch(Wp + (size_t)ctbase * 4 * 32 * 16, 0, 3);
    v16h a[4];
#pragma unroll
    for (int kt = 0; kt < 4; kt++) a[kt] = load_a_frag_f16(xn, HID, rt * 16, kt * 32);
    int l = threadIdx.x & 31, hsel = l >> 4;
#pragma unroll
    for (int c = 0; c < 4; c++) {
        v8f acc = {};
#pragma unroll
        for (int kt = 0; kt < 4; kt++)
            acc = wmma16(a[kt], load_b_packed(Wp, ctbase + c, kt, 4), acc);
        int n = (ctbase + c) * 16 + (l & 15);
#pragma unroll
        for (int i = 0; i < 8; i++)
            Out[(size_t)(rt * 16 + i + 8 * hsel) * HID + n] = acc[i];
    }
}

// -------------------- edge attention-bias MLP (ea path) --------------------
__global__ void k_edge_bias(const _Float16* __restrict__ eah,
                            const _Float16* __restrict__ w1p, const float* __restrict__ b1,
                            const _Float16* __restrict__ w2p, const float* __restrict__ b2,
                            float* __restrict__ bias_out) {
    __shared__ _Float16 lds[16 * 64];
    int e0 = blockIdx.x * 16;
    int l = threadIdx.x & 31, hsel = l >> 4, ncol = l & 15;
    v16h a = load_a_frag_f16(eah, 32, e0, 0);
#pragma unroll
    for (int c = 0; c < 4; c++) {
        v8f acc = {};
        acc = wmma16(a, load_b_packed(w1p, c, 0, 1), acc);
        float bb = b1[c * 16 + ncol];
#pragma unroll
        for (int i = 0; i < 8; i++)
            lds[(i + 8 * hsel) * 64 + c * 16 + ncol] = (_Float16)gelu_erf(acc[i] + bb);
    }
    __syncthreads();
    v8f acc = {};
#pragma unroll
    for (int kt = 0; kt < 2; kt++)
        acc = wmma16(load_a_frag_f16(lds, 64, 0, kt * 32),
                     load_b_packed(w2p, 0, kt, 2), acc);   // cols 8..15 packed as zero
    if (ncol < 8) {
        float bb = b2[ncol];
#pragma unroll
        for (int i = 0; i < 8; i++)
            bias_out[(size_t)(e0 + i + 8 * hsel) * 8 + ncol] = acc[i] + bb;
    }
}

// -------------------- per-edge logits + segment max --------------------
__global__ void k_logits(const float* __restrict__ Q, const float* __restrict__ K,
                         const int* __restrict__ src, const int* __restrict__ dst,
                         float* __restrict__ buf /* in: bias, out: logits */,
                         int* __restrict__ mbits) {
    int t = blockIdx.x * blockDim.x + threadIdx.x;
    if (t >= EE * 8) return;
    int e = t >> 3, h = t & 7;
    int s = src[e], d = dst[e];
    const float4* q4 = (const float4*)(Q + (size_t)d * HID + h * 16);
    const float4* k4 = (const float4*)(K + (size_t)s * HID + h * 16);
    float acc = 0.0f;
#pragma unroll
    for (int j = 0; j < 4; j++) {
        float4 qa = q4[j], ka = k4[j];
        acc += qa.x * ka.x + qa.y * ka.y + qa.z * ka.z + qa.w * ka.w;
    }
    float lg = acc * 0.25f + buf[t];
    buf[t] = lg;
    atomicMax(&mbits[(size_t)d * 8 + h], __float_as_int(lg));  // init 0 => clamp max at 0
}

// -------------------- exp + segment sum --------------------
__global__ void k_exp(const int* __restrict__ dst, const int* __restrict__ mbits,
                      float* __restrict__ buf /* in: logits, out: ex */,
                      float* __restrict__ ssum) {
    int t = blockIdx.x * blockDim.x + threadIdx.x;
    if (t >= EE * 8) return;
    int e = t >> 3, h = t & 7;
    int d = dst[e];
    float m = __int_as_float(mbits[(size_t)d * 8 + h]);
    float ex = expf(buf[t] - m);
    buf[t] = ex;
    atomicAdd(&ssum[(size_t)d * 8 + h], ex);
}

// -------------------- fused gate MLP (eg path) + weighted scatter-add --------------------
__global__ void k_agg(const _Float16* __restrict__ eah,
                      const _Float16* __restrict__ w1p, const float* __restrict__ b1,
                      const _Float16* __restrict__ w2p, const float* __restrict__ b2,
                      const float* __restrict__ V,
                      const int* __restrict__ src, const int* __restrict__ dst,
                      const float* __restrict__ ex, const float* __restrict__ ssum,
                      float* __restrict__ agg) {
    __shared__ _Float16 lds[16 * 64];
    int e0 = blockIdx.x * 16;
    int l = threadIdx.x & 31, hsel = l >> 4, ncol = l & 15;
    v16h a = load_a_frag_f16(eah, 32, e0, 0);
#pragma unroll
    for (int c = 0; c < 4; c++) {
        v8f acc = {};
        acc = wmma16(a, load_b_packed(w1p, c, 0, 1), acc);
        float bb = b1[c * 16 + ncol];
#pragma unroll
        for (int i = 0; i < 8; i++)
            lds[(i + 8 * hsel) * 64 + c * 16 + ncol] = (_Float16)gelu_erf(acc[i] + bb);
    }
    __syncthreads();
    v16h a2[2];
    a2[0] = load_a_frag_f16(lds, 64, 0, 0);
    a2[1] = load_a_frag_f16(lds, 64, 0, 32);
    int se[8], de[8];
#pragma unroll
    for (int i = 0; i < 8; i++) {
        int e = e0 + i + 8 * hsel;
        se[i] = src[e];
        de[i] = dst[e];
    }
#pragma unroll
    for (int h = 0; h < 8; h++) {           // one 16-col tile per head
        v8f acc = {};
#pragma unroll
        for (int kt = 0; kt < 2; kt++)
            acc = wmma16(a2[kt], load_b_packed(w2p, h, kt, 2), acc);
        float bb = b2[h * 16 + ncol];
#pragma unroll
        for (int i = 0; i < 8; i++) {
            int e = e0 + i + 8 * hsel;
            float gate = 1.0f / (1.0f + expf(-(acc[i] + bb)));
            float at = ex[(size_t)e * 8 + h] / (ssum[(size_t)de[i] * 8 + h] + 1e-10f);
            float val = at * gate * V[(size_t)se[i] * HID + h * 16 + ncol];
            atomicAdd(&agg[(size_t)de[i] * HID + h * 16 + ncol], val);
        }
    }
}

// -------------------- output projection + gamma/beta + residual --------------------
__global__ void k_outproj(const _Float16* __restrict__ aggh, const _Float16* __restrict__ Wop,
                          const float* __restrict__ bo, const float* __restrict__ gamma,
                          const float* __restrict__ beta, const float* __restrict__ x,
                          float* __restrict__ x1) {
    int rt = blockIdx.x * (blockDim.x >> 5) + (threadIdx.x >> 5);
    if (rt >= NROWT) return;
    int ctbase = blockIdx.y * 4;        // grid.y = 2
    __builtin_prefetch(Wop + (size_t)ctbase * 4 * 32 * 16, 0, 3);
    v16h a[4];
#pragma unroll
    for (int kt = 0; kt < 4; kt++) a[kt] = load_a_frag_f16(aggh, HID, rt * 16, kt * 32);
    int l = threadIdx.x & 31, hsel = l >> 4;
#pragma unroll
    for (int c = 0; c < 4; c++) {
        v8f acc = {};
#pragma unroll
        for (int kt = 0; kt < 4; kt++)
            acc = wmma16(a[kt], load_b_packed(Wop, ctbase + c, kt, 4), acc);
        int n = (ctbase + c) * 16 + (l & 15);
        float bias = bo[n];
#pragma unroll
        for (int i = 0; i < 8; i++) {
            size_t idx = (size_t)(rt * 16 + i + 8 * hsel) * HID + n;
            x1[idx] = x[idx] + gamma[idx] * (acc[i] + bias) + beta[idx];
        }
    }
}

// -------------------- FFN layer 1 (gelu -> f16) --------------------
__global__ void k_ffn1(const _Float16* __restrict__ xn2, const _Float16* __restrict__ w1p,
                       const float* __restrict__ b1, _Float16* __restrict__ hbuf) {
    int rt = blockIdx.x * (blockDim.x >> 5) + (threadIdx.x >> 5);
    if (rt >= NROWT) return;
    int ctbase = blockIdx.y * 4;        // grid.y = 4 -> 16 col tiles
    v16h a[4];
#pragma unroll
    for (int kt = 0; kt < 4; kt++) a[kt] = load_a_frag_f16(xn2, HID, rt * 16, kt * 32);
    int l = threadIdx.x & 31, hsel = l >> 4;
#pragma unroll
    for (int c = 0; c < 4; c++) {
        v8f acc = {};
#pragma unroll
        for (int kt = 0; kt < 4; kt++)
            acc = wmma16(a[kt], load_b_packed(w1p, ctbase + c, kt, 4), acc);
        int n = (ctbase + c) * 16 + (l & 15);
        float bias = b1[n];
#pragma unroll
        for (int i = 0; i < 8; i++)
            hbuf[(size_t)(rt * 16 + i + 8 * hsel) * 256 + n] = (_Float16)gelu_erf(acc[i] + bias);
    }
}

// -------------------- FFN layer 2 + residual -> d_out --------------------
__global__ void k_ffn2(const _Float16* __restrict__ hbuf, const _Float16* __restrict__ w2p,
                       const float* __restrict__ b2, const float* __restrict__ x1,
                       float* __restrict__ out) {
    int rt = blockIdx.x * (blockDim.x >> 5) + (threadIdx.x >> 5);
    if (rt >= NROWT) return;
    int ctbase = blockIdx.y * 4;        // grid.y = 2 -> 8 col tiles
    v16h a[8];
#pragma unroll
    for (int kt = 0; kt < 8; kt++) a[kt] = load_a_frag_f16(hbuf, 256, rt * 16, kt * 32);
    int l = threadIdx.x & 31, hsel = l >> 4;
#pragma unroll
    for (int c = 0; c < 4; c++) {
        v8f acc = {};
#pragma unroll
        for (int kt = 0; kt < 8; kt++)
            acc = wmma16(a[kt], load_b_packed(w2p, ctbase + c, kt, 8), acc);
        int n = (ctbase + c) * 16 + (l & 15);
        float bias = b2[n];
#pragma unroll
        for (int i = 0; i < 8; i++) {
            size_t idx = (size_t)(rt * 16 + i + 8 * hsel) * HID + n;
            out[idx] = x1[idx] + acc[i] + bias;
        }
    }
}

extern "C" void kernel_launch(void* const* d_in, const int* in_sizes, int n_in,
                              void* d_out, int out_size, void* d_ws, size_t ws_size,
                              hipStream_t stream) {
    const float* x      = (const float*)d_in[0];
    const float* ea     = (const float*)d_in[1];
    const float* gamma  = (const float*)d_in[2];
    const float* beta   = (const float*)d_in[3];
    const float* Wq     = (const float*)d_in[4];
    const float* Wk     = (const float*)d_in[5];
    const float* Wv     = (const float*)d_in[6];
    const float* Wo     = (const float*)d_in[7];
    const float* bo     = (const float*)d_in[8];
    const float* ea_w1  = (const float*)d_in[9];
    const float* ea_b1  = (const float*)d_in[10];
    const float* ea_w2  = (const float*)d_in[11];
    const float* ea_b2  = (const float*)d_in[12];
    const float* eg_w1  = (const float*)d_in[13];
    const float* eg_b1  = (const float*)d_in[14];
    const float* eg_w2  = (const float*)d_in[15];
    const float* eg_b2  = (const float*)d_in[16];
    const float* ln1_g  = (const float*)d_in[17];
    const float* ln1_b  = (const float*)d_in[18];
    const float* ln2_g  = (const float*)d_in[19];
    const float* ln2_b  = (const float*)d_in[20];
    const float* ffn_w1 = (const float*)d_in[21];
    const float* ffn_b1 = (const float*)d_in[22];
    const float* ffn_w2 = (const float*)d_in[23];
    const float* ffn_b2 = (const float*)d_in[24];
    const int*   eidx   = (const int*)d_in[25];
    const int* src = eidx;
    const int* dst = eidx + EE;
    float* out = (float*)d_out;

    // workspace carve-up (256B aligned)
    char* ws = (char*)d_ws;
    size_t off = 0;
    auto carve = [&](size_t bytes) { size_t o = off; off = (off + bytes + 255) & ~(size_t)255; return o; };
    _Float16* xn    = (_Float16*)(ws + carve((size_t)NN * HID * 2));
    float*    Qf    = (float*)   (ws + carve((size_t)NN * HID * 4));
    float*    Kf    = (float*)   (ws + carve((size_t)NN * HID * 4));
    float*    Vf    = (float*)   (ws + carve((size_t)NN * HID * 4));
    float*    ebuf  = (float*)   (ws + carve((size_t)EE * 8 * 4));   // bias -> logits -> ex (in place)
    int*      mb    = (int*)     (ws + carve((size_t)NN * 8 * 4));
    float*    ssum  = (float*)   (ws + carve((size_t)NN * 8 * 4));
    float*    aggb  = (float*)   (ws + carve((size_t)NN * HID * 4));
    _Float16* aggh  = (_Float16*)(ws + carve((size_t)NN * HID * 2));
    float*    x1    = (float*)   (ws + carve((size_t)NN * HID * 4));
    _Float16* xn2   = (_Float16*)(ws + carve((size_t)NN * HID * 2));
    _Float16* hbuf  = (_Float16*)(ws + carve((size_t)NN * 256 * 2));
    _Float16* eah   = (_Float16*)(ws + carve((size_t)EE * 32 * 2));
    // packed weights (f16, fragment order)
    _Float16* Wqp   = (_Float16*)(ws + carve((size_t)128 * 128 * 2));
    _Float16* Wkp   = (_Float16*)(ws + carve((size_t)128 * 128 * 2));
    _Float16* Wvp   = (_Float16*)(ws + carve((size_t)128 * 128 * 2));
    _Float16* Wop   = (_Float16*)(ws + carve((size_t)128 * 128 * 2));
    _Float16* f1p   = (_Float16*)(ws + carve((size_t)128 * 256 * 2));
    _Float16* f2p   = (_Float16*)(ws + carve((size_t)256 * 128 * 2));
    _Float16* ea1p  = (_Float16*)(ws + carve((size_t)32 * 64 * 2));
    _Float16* ea2p  = (_Float16*)(ws + carve((size_t)64 * 16 * 2));   // padded 8->16 cols
    _Float16* eg1p  = (_Float16*)(ws + carve((size_t)32 * 64 * 2));
    _Float16* eg2p  = (_Float16*)(ws + carve((size_t)64 * 128 * 2));

    hipMemsetAsync(mb,   0, (size_t)NN * 8 * 4,   stream);  // 0 bits == 0.0f => clamp max at 0
    hipMemsetAsync(ssum, 0, (size_t)NN * 8 * 4,   stream);
    hipMemsetAsync(aggb, 0, (size_t)NN * HID * 4, stream);

    // weight packing (grid = colTilesPad * ktiles)
    k_pack_w<<<dim3(8 * 4),  dim3(32), 0, stream>>>(Wq,     Wqp, 128, 128);
    k_pack_w<<<dim3(8 * 4),  dim3(32), 0, stream>>>(Wk,     Wkp, 128, 128);
    k_pack_w<<<dim3(8 * 4),  dim3(32), 0, stream>>>(Wv,     Wvp, 128, 128);
    k_pack_w<<<dim3(8 * 4),  dim3(32), 0, stream>>>(Wo,     Wop, 128, 128);
    k_pack_w<<<dim3(16 * 4), dim3(32), 0, stream>>>(ffn_w1, f1p, 128, 256);
    k_pack_w<<<dim3(8 * 8),  dim3(32), 0, stream>>>(ffn_w2, f2p, 256, 128);
    k_pack_w<<<dim3(4 * 1),  dim3(32), 0, stream>>>(ea_w1,  ea1p, 32, 64);
    k_pack_w<<<dim3(1 * 2),  dim3(32), 0, stream>>>(ea_w2,  ea2p, 64, 8);   // pad cols to 16
    k_pack_w<<<dim3(4 * 1),  dim3(32), 0, stream>>>(eg_w1,  eg1p, 32, 64);
    k_pack_w<<<dim3(8 * 2),  dim3(32), 0, stream>>>(eg_w2,  eg2p, 64, 128);

    // edge_attr -> f16
    k_cvt_f16<<<dim3((EE * 32 / 4 + 255) / 256), dim3(256), 0, stream>>>(ea, eah, EE * 32 / 4);

    const int nodeGemmGrid = (NROWT + 7) / 8;   // 8 waves per 256-thread block

    k_ln_f16<<<dim3((NN + 7) / 8), dim3(256), 0, stream>>>(x, ln1_g, ln1_b, xn, NN);
    k_qkv<<<dim3(nodeGemmGrid, 6), dim3(256), 0, stream>>>(xn, Wqp, Wkp, Wvp, Qf, Kf, Vf);
    k_edge_bias<<<dim3(NEDGET), dim3(32), 0, stream>>>(eah, ea1p, ea_b1, ea2p, ea_b2, ebuf);
    k_logits<<<dim3((EE * 8) / 256), dim3(256), 0, stream>>>(Qf, Kf, src, dst, ebuf, mb);
    k_exp<<<dim3((EE * 8) / 256), dim3(256), 0, stream>>>(dst, mb, ebuf, ssum);
    k_agg<<<dim3(NEDGET), dim3(32), 0, stream>>>(eah, eg1p, eg_b1, eg2p, eg_b2,
                                                 Vf, src, dst, ebuf, ssum, aggb);
    k_cvt_f16<<<dim3((NN * HID / 4 + 255) / 256), dim3(256), 0, stream>>>(aggb, aggh, NN * HID / 4);
    k_outproj<<<dim3(nodeGemmGrid, 2), dim3(256), 0, stream>>>(aggh, Wop, bo, gamma, beta, x, x1);
    k_ln_f16<<<dim3((NN + 7) / 8), dim3(256), 0, stream>>>(x1, ln2_g, ln2_b, xn2, NN);
    k_ffn1<<<dim3(nodeGemmGrid, 4), dim3(256), 0, stream>>>(xn2, f1p, ffn_b1, hbuf);
    k_ffn2<<<dim3(nodeGemmGrid, 2), dim3(256), 0, stream>>>(hbuf, f2p, ffn_b2, x1, out);
}
